// KDGCN_23235773071471
// MI455X (gfx1250) — matrix operand order repaired and checked by reading, not compiled, not run on hardware
//
#include <hip/hip_runtime.h>
#include <cstdint>
#include <cstddef>

// ---------------------------------------------------------------------------
// KDGCN attention + graph-conv for MI455X (gfx1250, wave32, WMMA bf16)
// Heavy GEMMs -> v_wmma_f32_16x16x32_bf16; projection GEMMs stage the shared
// A-panel through LDS with global_load_async_to_lds_b128 (ASYNCcnt), double
// buffered so the DMA overlaps the WMMAs.
// ---------------------------------------------------------------------------

typedef __attribute__((ext_vector_type(16))) __bf16 v16bf;
typedef __attribute__((ext_vector_type(8)))  float  v8f;
typedef __attribute__((ext_vector_type(4)))  unsigned int u32x4;

#define NB    32            // batches
#define NT    512           // tokens
#define ND    791           // feature dim
#define NH    7             // heads
#define DKH   113           // dk per head
#define DPAD  832           // D padded to mult of 64 (one b128/lane per async chunk)
#define HPAD  128           // per-head dk padded to mult of 32
#define DH    (NH * HPAD)   // 896: head-padded feature dim
#define MROWS (NB * NT)     // 16384 token rows

union FragU { v16bf v; u32x4 u[2]; };

// A-fragment (16x32 bf16, MxK): lane l holds row (base + (l&15));
// half-wave selects K-subgroup of 8: [k0+g0,+8) and [k0+g0+16,+8), g0=(l>>4)*8.
__device__ __forceinline__ v16bf load_frag_a(const __bf16* base, int ldk,
                                             int row_base, int k0) {
  int lane = threadIdx.x & 31;
  const __bf16* p = base + (size_t)(row_base + (lane & 15)) * ldk
                         + k0 + ((lane >> 4) << 3);
  FragU f;
  f.u[0] = *(const u32x4*)(p);
  f.u[1] = *(const u32x4*)(p + 16);
  return f.v;
}

// A-fragment from a staged LDS panel (32 rows x 64 cols, row-major)
__device__ __forceinline__ v16bf load_frag_a_lds(const __bf16* panel,
                                                 int waveM, int kc) {
  int lane = threadIdx.x & 31;
  const __bf16* p = panel + (waveM * 16 + (lane & 15)) * 64
                          + kc + ((lane >> 4) << 3);
  FragU f;
  f.u[0] = *(const u32x4*)(p);
  f.u[1] = *(const u32x4*)(p + 16);
  return f.v;
}

// B-fragment (32x16 bf16, KxN): lane l holds column (base_row + (l&15)) with
// K values [k0+g0,+8) and [k0+g0+16,+8), g0=(l>>4)*8 (source stored rowwise:
// row index = output column, contiguous K). Clamped for tail tiles.
__device__ __forceinline__ v16bf load_frag_b(const __bf16* base, int ldk,
                                             int row_base, int k0, int row_max) {
  int lane = threadIdx.x & 31;
  int r = row_base + (lane & 15);
  if (r >= row_max) r = row_max - 1;
  const __bf16* p = base + (size_t)r * ldk + k0 + ((lane >> 4) << 3);
  FragU f;
  f.u[0] = *(const u32x4*)(p);
  f.u[1] = *(const u32x4*)(p + 16);
  return f.v;
}

__device__ __forceinline__ v8f wmma_bf16(v16bf a, v16bf b, v8f c) {
  return __builtin_amdgcn_wmma_f32_16x16x32_bf16(false, a, false, b,
                                                 (short)0, c, false, false);
}

__device__ __forceinline__ v8f zero8() {
  v8f z = {0.f, 0.f, 0.f, 0.f, 0.f, 0.f, 0.f, 0.f};
  return z;
}

// CDNA5 async DMA: per-lane 16B global -> LDS, tracked by ASYNCcnt.
__device__ __forceinline__ void async_copy_b128(void* lds_dst, const void* gsrc) {
  asm volatile("global_load_async_to_lds_b128 %0, %1, off"
               :: "v"((unsigned int)(uintptr_t)lds_dst), "v"(gsrc)
               : "memory");
}
__device__ __forceinline__ void wait_async0() {
  asm volatile("s_wait_asynccnt 0" ::: "memory");
}

// ---------------------------------------------------------------------------
// Stage 0: conversions / fills
// ---------------------------------------------------------------------------
__global__ __launch_bounds__(256) void conv_fuse_kernel(
    const float* __restrict__ in, __bf16* __restrict__ out) {
  size_t i = (size_t)blockIdx.x * 256 + threadIdx.x;
  if (i >= (size_t)MROWS * DPAD) return;
  int c = (int)(i % DPAD);
  size_t r = i / DPAD;
  out[i] = (c < ND) ? (__bf16)in[r * ND + c] : (__bf16)0.0f;
}

// WT[n][k] = W[k][n], zero-padded to DPAD x DPAD
__global__ __launch_bounds__(256) void conv_wt_kernel(
    const float* __restrict__ W, __bf16* __restrict__ WT) {
  int i = blockIdx.x * 256 + threadIdx.x;
  if (i >= DPAD * DPAD) return;
  int k = i % DPAD, n = i / DPAD;
  WT[i] = (n < ND && k < ND) ? (__bf16)W[(size_t)k * ND + n] : (__bf16)0.0f;
}

__global__ __launch_bounds__(256) void fill_zero_kernel(
    unsigned int* __restrict__ p, size_t n) {
  size_t i = (size_t)blockIdx.x * 256 + threadIdx.x;
  if (i < n) p[i] = 0u;
}

// ---------------------------------------------------------------------------
// Staged GEMM core: all 8 waves of the block share a 32-row A panel. Chunks of
// 32x64 bf16 (4KB = 256 lanes x 16B) are DMA'd to LDS (double buffered) while
// WMMAs consume the previous chunk. One barrier per K-step.
// acc[j] += A[32xK] x B[Kx64-per-wave]
// ---------------------------------------------------------------------------
__device__ __forceinline__ void gemm_staged(
    const __bf16* __restrict__ A, const __bf16* __restrict__ Bsrc,
    int ldb, int browmax, __bf16 (*apanel)[32][64],
    int rowBlock, int waveM, int col0, v8f acc[4]) {
  int tid = threadIdx.x;
  int crow = tid >> 3, cseg = (tid & 7) << 3;   // 32 rows x 8 segs of 8 bf16
  const __bf16* asrc = A + (size_t)(rowBlock + crow) * DPAD + cseg;
  const int nk = DPAD / 64;                     // 13 steps
  async_copy_b128(&apanel[0][crow][cseg], asrc);
  for (int s = 0; s < nk; ++s) {
    wait_async0();        // our async portions of chunk s landed
    __syncthreads();      // everyone's chunk s visible; buf[(s+1)&1] idle
    if (s + 1 < nk)
      async_copy_b128(&apanel[(s + 1) & 1][crow][cseg], asrc + (s + 1) * 64);
    const __bf16* panel = &apanel[s & 1][0][0];
#pragma unroll
    for (int kc = 0; kc < 64; kc += 32) {
      v16bf af = load_frag_a_lds(panel, waveM, kc);
      int k0 = s * 64 + kc;
#pragma unroll
      for (int j = 0; j < 4; ++j) {
        v16bf bf = load_frag_b(Bsrc, ldb, col0 + j * 16, k0, browmax);
        acc[j] = wmma_bf16(af, bf, acc[j]);
      }
    }
  }
}

// ---------------------------------------------------------------------------
// Stage 1: projections. Block: 8 waves, tile 32x256 (wave tile 16x64).
// ---------------------------------------------------------------------------
__global__ __launch_bounds__(256) void gemm_qk_kernel(
    const __bf16* __restrict__ A, const __bf16* __restrict__ WT,
    const float* __restrict__ bias, __bf16* __restrict__ outH) {
  __shared__ __align__(16) __bf16 apanel[2][32][64];
  int wid = threadIdx.x >> 5, lane = threadIdx.x & 31;
  int waveM = wid & 1;
  int rowBlock = blockIdx.y * 32;
  int col0 = blockIdx.x * 256 + (wid >> 1) * 64;
  v8f acc[4];
#pragma unroll
  for (int j = 0; j < 4; ++j) acc[j] = zero8();
  gemm_staged(A, WT, DPAD, DPAD, apanel, rowBlock, waveM, col0, acc);
  int row0 = rowBlock + waveM * 16;
  int colL = lane & 15, rowHi = (lane >> 4) << 3;
#pragma unroll
  for (int j = 0; j < 4; ++j) {
    int c = col0 + j * 16 + colL;
    if (c < ND) {
      int h = c / DKH, rr = c - h * DKH;   // remap to head-padded column
      float bv = bias[c];
#pragma unroll
      for (int r = 0; r < 8; ++r) {
        int row = row0 + r + rowHi;
        outH[(size_t)row * DH + h * HPAD + rr] = (__bf16)(acc[j][r] + bv);
      }
    }
  }
}

// hidden = fuse @ W_gc, stored transposed: hidT[b][d][m] (contiguous in m)
__global__ __launch_bounds__(256) void gemm_hidden_kernel(
    const __bf16* __restrict__ A, const __bf16* __restrict__ WT,
    __bf16* __restrict__ hidT) {
  __shared__ __align__(16) __bf16 apanel[2][32][64];
  int wid = threadIdx.x >> 5, lane = threadIdx.x & 31;
  int waveM = wid & 1;
  int rowBlock = blockIdx.y * 32;
  int col0 = blockIdx.x * 256 + (wid >> 1) * 64;
  v8f acc[4];
#pragma unroll
  for (int j = 0; j < 4; ++j) acc[j] = zero8();
  gemm_staged(A, WT, DPAD, DPAD, apanel, rowBlock, waveM, col0, acc);
  int row0 = rowBlock + waveM * 16;
  int colL = lane & 15, rowHi = (lane >> 4) << 3;
#pragma unroll
  for (int j = 0; j < 4; ++j) {
    int c = col0 + j * 16 + colL;
    if (c < ND) {
#pragma unroll
      for (int r = 0; r < 8; ++r) {
        int row = row0 + r + rowHi;
        int b = row >> 9, m = row & (NT - 1);
        hidT[((size_t)b * DPAD + c) * NT + m] = (__bf16)acc[j][r];
      }
    }
  }
}

// ---------------------------------------------------------------------------
// Stage 2: fused gram + per-head scores + masked softmax + head-sum.
// One workgroup per (batch, 16-row block). Dynamic LDS:
//   sc[7][16][512] fp32 (224KB) + gr[16][512] fp32 (32KB) = 256KB (<=320KB/WGP)
// ---------------------------------------------------------------------------
__global__ __launch_bounds__(256) void attn_kernel(
    const __bf16* __restrict__ fuseB, const __bf16* __restrict__ Qh,
    const __bf16* __restrict__ Kh, float* __restrict__ attn) {
  extern __shared__ float lds[];
  float* sc = lds;                    // [NH][16][NT]
  float* gr = lds + NH * 16 * NT;     // [16][NT]
  int wid = threadIdx.x >> 5, lane = threadIdx.x & 31;
  int b = blockIdx.y, n0 = blockIdx.x * 16;
  int rowA = b * NT + n0;
  int colL = lane & 15, rowHi = (lane >> 4) << 3;

  for (int t = 0; t < 4; ++t) {
    int mt = wid + t * 8;             // 32 m-tiles split across 8 waves
    int rowB = b * NT + mt * 16;
    // gram tile (mask source), register double-buffered
    v8f g = zero8();
    v16bf ga = load_frag_a(fuseB, DPAD, rowA, 0);
    v16bf gb = load_frag_b(fuseB, DPAD, rowB, 0, MROWS);
    for (int k0 = 0; k0 < DPAD; k0 += 32) {
      int kn = (k0 + 32 < DPAD) ? (k0 + 32) : k0;
      v16bf ga2 = load_frag_a(fuseB, DPAD, rowA, kn);
      v16bf gb2 = load_frag_b(fuseB, DPAD, rowB, kn, MROWS);
      g = wmma_bf16(ga, gb, g);
      ga = ga2; gb = gb2;
    }
#pragma unroll
    for (int r = 0; r < 8; ++r)
      gr[(r + rowHi) * NT + mt * 16 + colL] = g[r];
    // per-head score tiles (head-padded K: 4 clean x32 chunks)
    for (int h = 0; h < NH; ++h) {
      v8f s = zero8();
#pragma unroll
      for (int kk = 0; kk < 4; ++kk) {
        int k0 = h * HPAD + kk * 32;
        s = wmma_bf16(load_frag_a(Qh, DH, rowA, k0),
                      load_frag_b(Kh, DH, rowB, k0, MROWS), s);
      }
#pragma unroll
      for (int r = 0; r < 8; ++r)
        sc[(h * 16 + r + rowHi) * NT + mt * 16 + colL] = s[r];
    }
  }
  __syncthreads();

  // masked softmax per (h, n) row: 8 waves x 14 rows = 112 rows, wave32
  const float rsdk = 0.09407208683f;  // 1/sqrt(113)
  for (int rowi = wid * 14; rowi < wid * 14 + 14; ++rowi) {
    int h = rowi >> 4, n = rowi & 15;
    float valid = (gr[n * NT] != 0.0f) ? 1.0f : 0.0f;
    float xs[16];
    float mx = -3.0e38f;
#pragma unroll
    for (int i = 0; i < 16; ++i) {
      int m = lane + i * 32;
      float gv = gr[n * NT + m];
      float x = (gv == 0.0f) ? -1.0e9f : sc[(h * 16 + n) * NT + m] * rsdk;
      xs[i] = x;
      mx = fmaxf(mx, x);
    }
#pragma unroll
    for (int o = 16; o > 0; o >>= 1) mx = fmaxf(mx, __shfl_xor(mx, o, 32));
    float sum = 0.f;
#pragma unroll
    for (int i = 0; i < 16; ++i) {
      float e = __expf(xs[i] - mx);
      xs[i] = e;
      sum += e;
    }
#pragma unroll
    for (int o = 16; o > 0; o >>= 1) sum += __shfl_xor(sum, o, 32);
    float inv = valid / sum;
#pragma unroll
    for (int i = 0; i < 16; ++i)
      sc[(h * 16 + n) * NT + lane + i * 32] = xs[i] * inv;
  }
  __syncthreads();

  // head-sum -> attn[b][n][m]
  for (int idx = threadIdx.x; idx < 16 * NT; idx += 256) {
    int n = idx >> 9, m = idx & (NT - 1);
    float s = 0.f;
#pragma unroll
    for (int h = 0; h < NH; ++h) s += sc[(h * 16 + n) * NT + m];
    attn[((size_t)(b * NT + n0 + n)) * NT + m] = s;
  }
}

// ---------------------------------------------------------------------------
// Stage 3a: per-batch top-2 over N*N -> kth[b] (the 2nd-largest value)
// ---------------------------------------------------------------------------
__global__ __launch_bounds__(256) void top2_kernel(
    const float* __restrict__ attn, float* __restrict__ kth) {
  int b = blockIdx.x;
  const float* a = attn + (size_t)b * NT * NT;
  float v1 = -3.0e38f, v2 = -3.0e38f;
  for (int i = threadIdx.x; i < NT * NT; i += 256) {
    float x = a[i];
    if (x > v1) { v2 = v1; v1 = x; }
    else if (x > v2) { v2 = x; }
  }
  __shared__ float s1[256], s2[256];
  s1[threadIdx.x] = v1; s2[threadIdx.x] = v2;
  __syncthreads();
  for (int off = 128; off > 0; off >>= 1) {
    if (threadIdx.x < (unsigned)off) {
      float a1 = s1[threadIdx.x], a2 = s2[threadIdx.x];
      float b1 = s1[threadIdx.x + off], b2 = s2[threadIdx.x + off];
      if (a1 >= b1) { s1[threadIdx.x] = a1; s2[threadIdx.x] = fmaxf(a2, b1); }
      else          { s1[threadIdx.x] = b1; s2[threadIdx.x] = fmaxf(b2, a1); }
    }
    __syncthreads();
  }
  if (threadIdx.x == 0) kth[b] = s2[0];
}

// ---------------------------------------------------------------------------
// Stage 3b: att_adj = (mask + mask^T, diag=1) * attn, bf16; + rowsum -> denom
// ---------------------------------------------------------------------------
__global__ __launch_bounds__(256) void adj_kernel(
    const float* __restrict__ attn, const float* __restrict__ kth,
    __bf16* __restrict__ adjB, float* __restrict__ denom) {
  int b = blockIdx.y, n = blockIdx.x;
  const float* A = attn + (size_t)b * NT * NT;
  float th = kth[b];
  float sum = 0.f;
  for (int m = threadIdx.x; m < NT; m += 256) {
    float a = A[n * NT + m];
    float at = A[m * NT + n];
    float w = (m == n) ? 1.0f
                       : ((a >= th ? 1.0f : 0.0f) + (at >= th ? 1.0f : 0.0f));
    float v = w * a;
    adjB[(size_t)b * NT * NT + n * NT + m] = (__bf16)v;
    sum += v;
  }
  __shared__ float red[256];
  red[threadIdx.x] = sum;
  __syncthreads();
  for (int off = 128; off > 0; off >>= 1) {
    if (threadIdx.x < (unsigned)off) red[threadIdx.x] += red[threadIdx.x + off];
    __syncthreads();
  }
  if (threadIdx.x == 0) denom[b * NT + n] = red[0] + 1.0f;
}

// ---------------------------------------------------------------------------
// Stage 4: out = relu( (adj @ hidden) / denom + b_gc ), per batch WMMA GEMM
// with register double-buffering across K-steps.
// ---------------------------------------------------------------------------
__global__ __launch_bounds__(256) void gemm_out_kernel(
    const __bf16* __restrict__ adjB, const __bf16* __restrict__ hidT,
    const float* __restrict__ denom, const float* __restrict__ bgc,
    float* __restrict__ out) {
  int wid = threadIdx.x >> 5, lane = threadIdx.x & 31;
  int b = blockIdx.z;
  int row0 = blockIdx.y * 32 + (wid & 1) * 16;
  int col0 = blockIdx.x * 256 + (wid >> 1) * 64;
  if (col0 >= ND) return;
  const __bf16* A = adjB + (size_t)b * NT * NT;
  const __bf16* Bm = hidT + (size_t)b * DPAD * NT;
  v8f acc[4];
#pragma unroll
  for (int j = 0; j < 4; ++j) acc[j] = zero8();
  v16bf a0 = load_frag_a(A, NT, row0, 0);
  v16bf b0[4];
#pragma unroll
  for (int j = 0; j < 4; ++j) b0[j] = load_frag_b(Bm, NT, col0 + j * 16, 0, DPAD);
  for (int k0 = 0; k0 < NT; k0 += 32) {
    int kn = (k0 + 32 < NT) ? (k0 + 32) : k0;
    v16bf a1 = load_frag_a(A, NT, row0, kn);
    v16bf b1[4];
#pragma unroll
    for (int j = 0; j < 4; ++j) b1[j] = load_frag_b(Bm, NT, col0 + j * 16, kn, DPAD);
#pragma unroll
    for (int j = 0; j < 4; ++j) acc[j] = wmma_bf16(a0, b0[j], acc[j]);
    a0 = a1;
#pragma unroll
    for (int j = 0; j < 4; ++j) b0[j] = b1[j];
  }
  int colL = lane & 15, rowHi = (lane >> 4) << 3;
#pragma unroll
  for (int j = 0; j < 4; ++j) {
    int c = col0 + j * 16 + colL;
    if (c < ND) {
      float bv = bgc[c];
#pragma unroll
      for (int r = 0; r < 8; ++r) {
        int row = row0 + r + rowHi;
        float v = acc[j][r] / denom[b * NT + row] + bv;
        out[((size_t)b * NT + row) * ND + c] = fmaxf(v, 0.0f);
      }
    }
  }
}

// ---------------------------------------------------------------------------
// Host launcher
// ---------------------------------------------------------------------------
extern "C" void kernel_launch(void* const* d_in, const int* in_sizes, int n_in,
                              void* d_out, int out_size, void* d_ws,
                              size_t ws_size, hipStream_t stream) {
  (void)in_sizes; (void)n_in; (void)out_size; (void)ws_size;
  const float* fuse = (const float*)d_in[0];
  const float* Wq   = (const float*)d_in[1];
  const float* bq   = (const float*)d_in[2];
  const float* Wk   = (const float*)d_in[3];
  const float* bk   = (const float*)d_in[4];
  const float* Wgc  = (const float*)d_in[5];
  const float* bgc  = (const float*)d_in[6];
  float* out = (float*)d_out;

  char* p = (char*)d_ws;
  auto alloc = [&](size_t bytes) {
    char* r = p;
    p += (bytes + 255) & ~(size_t)255;
    return r;
  };
  __bf16* fuseB = (__bf16*)alloc((size_t)MROWS * DPAD * 2);
  __bf16* WqT   = (__bf16*)alloc((size_t)DPAD * DPAD * 2);
  __bf16* WkT   = (__bf16*)alloc((size_t)DPAD * DPAD * 2);
  __bf16* WgT   = (__bf16*)alloc((size_t)DPAD * DPAD * 2);
  __bf16* Qh    = (__bf16*)alloc((size_t)MROWS * DH * 2);
  __bf16* Kh    = (__bf16*)alloc((size_t)MROWS * DH * 2);
  __bf16* hidT  = (__bf16*)alloc((size_t)NB * DPAD * NT * 2);
  float*  attn  = (float*)alloc((size_t)NB * NT * NT * 4);
  __bf16* adjB  = (__bf16*)alloc((size_t)NB * NT * NT * 2);
  float*  denom = (float*)alloc((size_t)MROWS * 4);
  float*  kth   = (float*)alloc((size_t)NB * 4);

  // Stage 0: convert + zero pads
  {
    size_t n = (size_t)MROWS * DPAD;
    conv_fuse_kernel<<<dim3((unsigned)((n + 255) / 256)), 256, 0, stream>>>(fuse, fuseB);
    unsigned wblocks = (DPAD * DPAD + 255) / 256;
    conv_wt_kernel<<<wblocks, 256, 0, stream>>>(Wq, WqT);
    conv_wt_kernel<<<wblocks, 256, 0, stream>>>(Wk, WkT);
    conv_wt_kernel<<<wblocks, 256, 0, stream>>>(Wgc, WgT);
    size_t qdw = (size_t)MROWS * DH / 2;           // bytes/4
    fill_zero_kernel<<<dim3((unsigned)((qdw + 255) / 256)), 256, 0, stream>>>((unsigned int*)Qh, qdw);
    fill_zero_kernel<<<dim3((unsigned)((qdw + 255) / 256)), 256, 0, stream>>>((unsigned int*)Kh, qdw);
    size_t hdw = (size_t)NB * DPAD * NT / 2;
    fill_zero_kernel<<<dim3((unsigned)((hdw + 255) / 256)), 256, 0, stream>>>((unsigned int*)hidT, hdw);
  }

  // Stage 1: projections (WMMA bf16, async-LDS staged A panel)
  dim3 pgrid(4, MROWS / 32);
  gemm_qk_kernel<<<pgrid, 256, 0, stream>>>(fuseB, WqT, bq, Qh);
  gemm_qk_kernel<<<pgrid, 256, 0, stream>>>(fuseB, WkT, bk, Kh);
  gemm_hidden_kernel<<<pgrid, 256, 0, stream>>>(fuseB, WgT, hidT);

  // Stage 2: fused gram/scores/softmax/head-sum (256KB dynamic LDS)
  size_t lds_bytes = (size_t)(NH * 16 * NT + 16 * NT) * 4;  // 262144
  attn_kernel<<<dim3(NT / 16, NB), 256, lds_bytes, stream>>>(fuseB, Qh, Kh, attn);

  // Stage 3: top-2 threshold, adjacency + rowsum
  top2_kernel<<<NB, 256, 0, stream>>>(attn, kth);
  adj_kernel<<<dim3(NT, NB), 256, 0, stream>>>(attn, kth, adjB, denom);

  // Stage 4: aggregation GEMM + epilogue
  gemm_out_kernel<<<dim3(4, NT / 32, NB), 256, 0, stream>>>(adjB, hidT, denom, bgc, out);
}
